// SparseGraphWaveletLayer_3831110828256
// MI455X (gfx1250) — compile-verified, matrix operand order checked by reading, not run on hardware
//
#include <hip/hip_runtime.h>

#define C_DIM 128
#define NTHREADS 256

typedef __attribute__((ext_vector_type(16))) _Float16 v16h;
typedef __attribute__((ext_vector_type(8)))  float    v8f;

// ---------------------------------------------------------------- utilities
__global__ void zero_f32(float4* __restrict__ p, int n4) {
    int i = blockIdx.x * blockDim.x + threadIdx.x;
    if (i < n4) p[i] = make_float4(0.f, 0.f, 0.f, 0.f);
}

__global__ void relu_f32(float4* __restrict__ p, int n4) {
    int i = blockIdx.x * blockDim.x + threadIdx.x;
    if (i < n4) {
        float4 v = p[i];
        v.x = fmaxf(v.x, 0.f); v.y = fmaxf(v.y, 0.f);
        v.z = fmaxf(v.z, 0.f); v.w = fmaxf(v.w, 0.f);
        p[i] = v;
    }
}

// ------------------------------------------------- densify feature matrix
// A[r, c] += v  : one f32 atomic per nnz (vs 128 in the naive SpMM path).
__global__ void scatter_coo(const int* __restrict__ rows,
                            const int* __restrict__ cols,
                            const float* __restrict__ vals,
                            float* __restrict__ A, int nnz) {
    int e = blockIdx.x * blockDim.x + threadIdx.x;
    if (e >= nnz) return;
    size_t idx = (size_t)rows[e] * C_DIM + (size_t)cols[e];
    __hip_atomic_fetch_add(&A[idx], vals[e], __ATOMIC_RELAXED,
                           __HIP_MEMORY_SCOPE_AGENT);
}

// ------------------------------------------- pre-swizzle W into B-fragments
// 32 fragments (4 K-chunks x 8 col tiles), 512 halves each, laid out so a
// lane's 16 B-matrix elements are one contiguous 32-byte block.
// ISA 7.12.2 16-bit operand layout: lanes 0-15 / 16-31 split the K-halves,
// VGPR j holds K pair  (j<4 ? kh*8+2j : 16+kh*8+2(j-4)).
__global__ void build_wfrag(const float* __restrict__ W,
                            _Float16* __restrict__ Bfrag) {
    int t = blockIdx.x * blockDim.x + threadIdx.x;   // 32*512 = 16384 threads
    if (t >= 32 * 512) return;
    int frag = t >> 9;          // 0..31
    int rem  = t & 511;
    int lane = rem >> 4;        // 0..31
    int e    = rem & 15;        // element within lane
    int k0   = (frag >> 3) * 32;
    int c    = frag & 7;
    int j = e >> 1, p = e & 1;
    int kh = lane >> 4;
    int kb = k0 + ((j < 4) ? (kh * 8 + 2 * j) : (16 + kh * 8 + 2 * (j - 4))) + p;
    int n  = c * 16 + (lane & 15);
    Bfrag[t] = (_Float16)W[kb * C_DIM + n];
}

// --------------------------------------------------- dense GEMM via WMMA
// filtered(N x 128) = A(N x 128, f32 -> f16) @ W(128 x 128, pre-swizzled f16)
// One wave per 16-row tile; 8 column tiles of 16; K in 4 chunks of 32.
__global__ void gemm_wmma_f16(const float* __restrict__ A,
                              const _Float16* __restrict__ Bfrag,
                              float* __restrict__ D, int nrows) {
    int gtid = blockIdx.x * blockDim.x + threadIdx.x;
    int wave = gtid >> 5;
    int lane = threadIdx.x & 31;
    int m0 = wave * 16;
    if (m0 >= nrows) return;                 // uniform per wave: EXEC all-ones

    int mrow = m0 + (lane & 15);
    int kh   = lane >> 4;

    v8f acc[8] = {};
#pragma unroll
    for (int kc = 0; kc < 4; ++kc) {
        int k0 = kc * 32;
        // A fragment: 8 contiguous f32 pairs per lane, convert to f16
        v16h a;
#pragma unroll
        for (int j = 0; j < 8; ++j) {
            int kb = k0 + ((j < 4) ? (kh * 8 + 2 * j)
                                   : (16 + kh * 8 + 2 * (j - 4)));
            const float2 f = *(const float2*)(A + (size_t)mrow * C_DIM + kb);
            a[2 * j]     = (_Float16)f.x;
            a[2 * j + 1] = (_Float16)f.y;
        }
#pragma unroll
        for (int c = 0; c < 8; ++c) {
            const v16h b = *(const v16h*)(Bfrag + ((size_t)(kc * 8 + c) * 512)
                                                + (size_t)lane * 16);
            acc[c] = __builtin_amdgcn_wmma_f32_16x16x32_f16(
                false, a, false, b, (short)0, acc[c], false, false);
        }
    }
    // C/D layout: VGPR v -> row m0 + kh*8 + v, col = tile*16 + lane%16
#pragma unroll
    for (int c = 0; c < 8; ++c) {
        int col = c * 16 + (lane & 15);
#pragma unroll
        for (int v = 0; v < 8; ++v) {
            int row = m0 + kh * 8 + v;
            D[(size_t)row * C_DIM + col] = acc[c][v];
        }
    }
}

// ------------------------------------------------- gather/scatter SpMM
// One wave per nnz; lane handles 4 channels (float4 gather = 512B/wave,
// fully coalesced; dense operands are L2-resident). 4 f32 atomics per lane.
// Optional diag fuses the row-diagonal Theta scale at gather time.
__global__ void spmm_scatter(const int* __restrict__ rows,
                             const int* __restrict__ cols,
                             const float* __restrict__ vals,
                             const float* __restrict__ diag,
                             const float* __restrict__ src,
                             float* __restrict__ dst, int nnz) {
    int gtid = blockIdx.x * blockDim.x + threadIdx.x;
    int e    = gtid >> 5;
    int lane = threadIdx.x & 31;
    if (e >= nnz) return;
    int r = rows[e];
    int c = cols[e];
    float v = vals[e];
    if (diag) v *= diag[c];
    const float4 s = *(const float4*)(src + (size_t)c * C_DIM + lane * 4);
    float* d = dst + (size_t)r * C_DIM + lane * 4;
    __hip_atomic_fetch_add(d + 0, s.x * v, __ATOMIC_RELAXED, __HIP_MEMORY_SCOPE_AGENT);
    __hip_atomic_fetch_add(d + 1, s.y * v, __ATOMIC_RELAXED, __HIP_MEMORY_SCOPE_AGENT);
    __hip_atomic_fetch_add(d + 2, s.z * v, __ATOMIC_RELAXED, __HIP_MEMORY_SCOPE_AGENT);
    __hip_atomic_fetch_add(d + 3, s.w * v, __ATOMIC_RELAXED, __HIP_MEMORY_SCOPE_AGENT);
}

// ---------------------------------------------------------------- driver
extern "C" void kernel_launch(void* const* d_in, const int* in_sizes, int n_in,
                              void* d_out, int out_size, void* d_ws, size_t ws_size,
                              hipStream_t stream) {
    const int*   phi_rows  = (const int*)d_in[0];
    const float* phi_vals  = (const float*)d_in[1];
    const int*   phii_rows = (const int*)d_in[2];
    const float* phii_vals = (const float*)d_in[3];
    const int*   f_rows    = (const int*)d_in[4];
    const float* f_vals    = (const float*)d_in[5];
    const float* W         = (const float*)d_in[6];
    const float* diag      = (const float*)d_in[7];
    // d_in[8] = dropout (0) -> identity at inference; ignored.

    const int nnz = in_sizes[1];     // phi_values count
    const int N   = in_sizes[7];     // diagonal_weight_filter rows

    const int* phi_cols  = phi_rows  + nnz;
    const int* phii_cols = phii_rows + nnz;
    const int* f_cols    = f_rows    + nnz;

    float*    A     = (float*)d_ws;  // N x 128 f32 panel (A, then z)
    _Float16* Bfrag = (_Float16*)((char*)d_ws + (size_t)N * C_DIM * sizeof(float));
    float*    out   = (float*)d_out;

    const int n4       = N * C_DIM / 4;
    const int zb       = (n4 + NTHREADS - 1) / NTHREADS;
    const int sb       = (nnz + NTHREADS - 1) / NTHREADS;
    const int spmm_b   = (int)(((size_t)nnz * 32 + NTHREADS - 1) / NTHREADS);
    const int gemm_b   = (int)((((size_t)((N + 15) / 16)) * 32 + NTHREADS - 1) / NTHREADS);

    // 1) densify feature COO into A
    zero_f32<<<zb, NTHREADS, 0, stream>>>((float4*)A, n4);
    scatter_coo<<<sb, NTHREADS, 0, stream>>>(f_rows, f_cols, f_vals, A, nnz);

    // 2) filtered = A @ W  (WMMA), result staged in d_out
    build_wfrag<<<(32 * 512) / NTHREADS, NTHREADS, 0, stream>>>(W, Bfrag);
    gemm_wmma_f16<<<gemm_b, NTHREADS, 0, stream>>>(A, Bfrag, out, N);

    // 3) z = Phi_inv @ filtered   (z into ws, filtered gathered from d_out)
    zero_f32<<<zb, NTHREADS, 0, stream>>>((float4*)A, n4);
    spmm_scatter<<<spmm_b, NTHREADS, 0, stream>>>(phii_rows, phii_cols, phii_vals,
                                                  nullptr, out, A, nnz);

    // 4) out = Phi @ (diag * z)   (diag fused into gather scale)
    zero_f32<<<zb, NTHREADS, 0, stream>>>((float4*)out, n4);
    spmm_scatter<<<spmm_b, NTHREADS, 0, stream>>>(phi_rows, phi_cols, phi_vals,
                                                  diag, A, out, nnz);

    // 5) relu in place
    relu_f32<<<zb, NTHREADS, 0, stream>>>((float4*)out, n4);
}
